// DeepSeekMLA_23433341567239
// MI455X (gfx1250) — matrix operand order repaired and checked by reading, not compiled, not run on hardware
//
#include <hip/hip_runtime.h>

typedef __attribute__((ext_vector_type(16))) __bf16 v16bf;
typedef __attribute__((ext_vector_type(8)))  float  v8f;

#define HIDDEN 2048
#define NHEADS 16
#define HEADD  128
#define RANK   64
#define SEQ    2048
#define BATCH  2

static __device__ __forceinline__ unsigned short f2bf(float f) {
  union { float f; unsigned u; } v; v.f = f;
  unsigned r = (v.u + 0x7FFFu + ((v.u >> 16) & 1u)) >> 16;
  return (unsigned short)r;
}
static __device__ __forceinline__ float bf2f(unsigned short h) {
  union { unsigned u; float f; } v; v.u = ((unsigned)h) << 16;
  return v.f;
}

union FragAB { v16bf v; unsigned u[8]; };

// CDNA5 async copy: 16B from global to LDS, tracked by ASYNCcnt (ISA 08_async_tensor).
static __device__ __forceinline__ void async_cp16(unsigned ldsOff, const void* g) {
  asm volatile("global_load_async_to_lds_b128 %0, %1, off"
               :: "v"(ldsOff), "v"(g) : "memory");
}
static __device__ __forceinline__ unsigned lds_off(const void* p) {
  // generic pointer to __shared__: addr[31:0] is the LDS byte offset (ISA 10.2)
  return (unsigned)(unsigned long long)p;
}
static __device__ __forceinline__ void wait_async0() {
  asm volatile("s_wait_asynccnt 0x0" ::: "memory");
}
static __device__ __forceinline__ void wait_async4() {
  asm volatile("s_wait_asynccnt 0x4" ::: "memory");
}

// ---------------------------------------------------------------- fp32->bf16
__global__ void cvt_f32_bf16(const float* __restrict__ src,
                             unsigned short* __restrict__ dst, int n) {
  int i = blockIdx.x * blockDim.x + threadIdx.x;
  int st = gridDim.x * blockDim.x;
  for (; i < n; i += st) dst[i] = f2bf(src[i]);
}

// fp32 (K,N) -> bf16 transposed (N,K); LDS-tiled so both sides stay coalesced.
__global__ __launch_bounds__(256)
void cvt_t_f32_bf16(const float* __restrict__ src, unsigned short* __restrict__ dst,
                    int K, int N) {
  __shared__ unsigned short t[32][33];
  const int kb = blockIdx.y * 32, nb = blockIdx.x * 32;
  const int tx = threadIdx.x & 31, ty = threadIdx.x >> 5;   // ty: 0..7
#pragma unroll
  for (int i = 0; i < 32; i += 8)
    t[ty + i][tx] = f2bf(src[(size_t)(kb + ty + i) * N + nb + tx]);
  __syncthreads();
#pragma unroll
  for (int i = 0; i < 32; i += 8)
    dst[(size_t)(nb + ty + i) * K + kb + tx] = t[tx][ty + i];
}

// ---------------------------------------------------------------- WMMA GEMM
// C[M,N] = A[M,K] * BT[N,K]^T; A row-major, BT = transposed weight (N,K), bf16.
// Tile BM=128, BN=128, BK=32; 256 threads = 8 waves (4 M x 2 N).
// Both tiles (identical 128x32 shape) staged with global_load_async_to_lds_b128,
// double buffered: 4 chunks/lane/tile -> s_wait_asynccnt 4 overlaps the next
// tile's copies with this tile's wmma. Fragment reads are 16B/32B contiguous
// (stride 40 u16 = 80B: 16B aligned, 20-dword bank stride = conflict-free).
#define LDT_S 40

template <int OUT_BF16>
__global__ __launch_bounds__(256)
void wmma_gemm(const unsigned short* __restrict__ A,
               const unsigned short* __restrict__ BT,
               void* __restrict__ Cout, int M, int N, int K) {
  __shared__ unsigned short Al[2][128 * LDT_S];  // (m, k)
  __shared__ unsigned short Bl[2][128 * LDT_S];  // (n, k)
  const int tid = threadIdx.x;
  const int lane = tid & 31;
  const int w = tid >> 5;
  const int wm = w & 3, wn = w >> 2;
  const int half = lane >> 4;
  const int l16 = lane & 15;
  const int rowBase = blockIdx.y * 128;
  const int colBase = blockIdx.x * 128;

  // per-thread async chunk coordinates: 512 16B chunks per tile, 2 per thread
  const int r0 = tid >> 2, c0 = (tid & 3) * 8;
  const int r1 = (tid + 256) >> 2, c1 = c0;

  auto issue = [&](int kb, int buf) {
    async_cp16(lds_off(&Al[buf][r0 * LDT_S + c0]),
               A + (size_t)(rowBase + r0) * K + kb + c0);
    async_cp16(lds_off(&Al[buf][r1 * LDT_S + c1]),
               A + (size_t)(rowBase + r1) * K + kb + c1);
    async_cp16(lds_off(&Bl[buf][r0 * LDT_S + c0]),
               BT + (size_t)(colBase + r0) * K + kb + c0);
    async_cp16(lds_off(&Bl[buf][r1 * LDT_S + c1]),
               BT + (size_t)(colBase + r1) * K + kb + c1);
  };

  v8f zero = {};
  v8f c[2][4];
#pragma unroll
  for (int mi = 0; mi < 2; ++mi)
#pragma unroll
    for (int ni = 0; ni < 4; ++ni) c[mi][ni] = zero;

  const int ntiles = K >> 5;
  issue(0, 0);
  for (int j = 0; j < ntiles; ++j) {
    const int buf = j & 1;
    if (j + 1 < ntiles) {
      issue((j + 1) << 5, buf ^ 1);
      wait_async4();           // tile j resident (async loads complete in order)
    } else {
      wait_async0();
    }
    __syncthreads();

    FragAB a[2], b[4];
#pragma unroll
    for (int mi = 0; mi < 2; ++mi) {
      int rt = wm * 32 + mi * 16 + l16;
#pragma unroll
      for (int i = 0; i < 8; ++i) {
        int k0 = (i < 4 ? 2 * i : 16 + 2 * (i - 4)) + 8 * half;  // ISA A layout
        a[mi].u[i] = *(const unsigned*)(&Al[buf][rt * LDT_S + k0]);
      }
    }
#pragma unroll
    for (int ni = 0; ni < 4; ++ni) {
      int ct = wn * 64 + ni * 16 + l16;
#pragma unroll
      for (int i = 0; i < 8; ++i) {
        int k0 = (i < 4 ? 2 * i : 16 + 2 * (i - 4)) + 8 * half;  // ISA B layout
        b[ni].u[i] = *(const unsigned*)(&Bl[buf][ct * LDT_S + k0]);
      }
    }
#pragma unroll
    for (int mi = 0; mi < 2; ++mi)
#pragma unroll
      for (int ni = 0; ni < 4; ++ni)
        c[mi][ni] = __builtin_amdgcn_wmma_f32_16x16x32_bf16(
            false, a[mi].v, false, b[ni].v, (short)0, c[mi][ni], false, false);
    __syncthreads();
  }

#pragma unroll
  for (int mi = 0; mi < 2; ++mi)
#pragma unroll
    for (int ni = 0; ni < 4; ++ni)
#pragma unroll
      for (int r = 0; r < 8; ++r) {
        int row = rowBase + wm * 32 + mi * 16 + half * 8 + r;     // ISA C layout
        int col = colBase + wn * 64 + ni * 16 + l16;
        if (col < N) {
          float v = c[mi][ni][r];
          if (OUT_BF16)
            ((unsigned short*)Cout)[(size_t)row * N + col] = f2bf(v);
          else
            ((float*)Cout)[(size_t)row * N + col] = v;
        }
      }
}

// ---------------------------------------------------------------- RoPE (in place)
__global__ void rope_kernel(unsigned short* __restrict__ Q,
                            unsigned short* __restrict__ Kt,
                            const int* __restrict__ pos_ids) {
  int gid = blockIdx.x * blockDim.x + threadIdx.x;
  int row = gid >> 5;               // (b*S+s)*16+h
  int t = gid & 31;
  int s = (row >> 4) & (SEQ - 1);
  float pos = (float)pos_ids[s];
  float inv = __powf(10000.f, -(float)t / 32.f);
  float f = pos * inv;
  float sn, cs;
  __sincosf(f, &sn, &cs);
  size_t base = (size_t)row * HEADD;
  float x1 = bf2f(Q[base + t]), x2 = bf2f(Q[base + t + 32]);
  Q[base + t]      = f2bf(x1 * cs - x2 * sn);
  Q[base + t + 32] = f2bf(x2 * cs + x1 * sn);
  x1 = bf2f(Kt[base + t]); x2 = bf2f(Kt[base + t + 32]);
  Kt[base + t]      = f2bf(x1 * cs - x2 * sn);
  Kt[base + t + 32] = f2bf(x2 * cs + x1 * sn);
}

// ---------------------------------------------------------------- Flash attention
// Per (b,h): softmax(Qa Kc^T / 8) Vc, rank 64, 32-key blocks.
// K tile async-copied to LDS; V transposed synchronously.
__global__ __launch_bounds__(256)
void flash_attn(const unsigned short* __restrict__ Qa,
                const unsigned short* __restrict__ Kc,
                const unsigned short* __restrict__ Vc,
                unsigned short* __restrict__ Oc) {
  __shared__ unsigned short Kl[32 * 72];        // (key, r), 144B rows: 16B aligned
  __shared__ unsigned short Vl[64 * 34];        // (r, key) transposed
  __shared__ unsigned short Pl[8 * 16 * 34];    // per-wave P staging
  const int tid = threadIdx.x, lane = tid & 31, w = tid >> 5;
  const int half = lane >> 4, l16 = lane & 15;
  const int bh = blockIdx.y;
  const int b = bh >> 4, h = bh & 15;
  const int qw = blockIdx.x * 128 + w * 16;

  // Q A-fragments (16 rows x 64 rank = two 16x32 frags), loaded once
  FragAB aq[2];
#pragma unroll
  for (int kk = 0; kk < 2; ++kk)
#pragma unroll
    for (int i = 0; i < 8; ++i) {
      int k0 = (i < 4 ? 2 * i : 16 + 2 * (i - 4)) + 8 * half + kk * 32;
      size_t g = ((size_t)((b * SEQ + qw + l16) * NHEADS + h)) * RANK + k0;
      aq[kk].u[i] = *(const unsigned*)(Qa + g);
    }

  float m[8], l[8];
  v8f zero = {};
  v8f o[4];
#pragma unroll
  for (int r = 0; r < 8; ++r) { m[r] = -1e30f; l[r] = 0.f; }
#pragma unroll
  for (int n = 0; n < 4; ++n) o[n] = zero;

  const int kkey = tid >> 3, kc16 = (tid & 7) * 8;   // K async chunk coords

  for (int kb = 0; kb < SEQ; kb += 32) {
    async_cp16(lds_off(&Kl[kkey * 72 + kc16]),
               Kc + ((size_t)((b * SEQ + kb + kkey) * NHEADS + h)) * RANK + kc16);
#pragma unroll
    for (int i = 0; i < 8; ++i) {
      int e = tid + i * 256;
      int r = e & 63, key = e >> 6;
      Vl[r * 34 + key] =
          Vc[((size_t)((b * SEQ + kb + key) * NHEADS + h)) * RANK + r];
    }
    wait_async0();
    __syncthreads();

    // scores: two 16-key tiles, rank 64 as two wmma each
    v8f s0 = zero, s1 = zero;
#pragma unroll
    for (int nt = 0; nt < 2; ++nt) {
      v8f acc = zero;
#pragma unroll
      for (int kk = 0; kk < 2; ++kk) {
        FragAB bb;
#pragma unroll
        for (int i = 0; i < 8; ++i) {
          int rr = kk * 32 + (i < 4 ? 2 * i : 16 + 2 * (i - 4)) + 8 * half;
          bb.u[i] = *(const unsigned*)(&Kl[(nt * 16 + l16) * 72 + rr]);
        }
        acc = __builtin_amdgcn_wmma_f32_16x16x32_bf16(
            false, aq[kk].v, false, bb.v, (short)0, acc, false, false);
      }
      if (nt == 0) s0 = acc; else s1 = acc;
    }

    // online softmax: rows live across one 16-lane half (ISA C layout)
    const float scale = 0.125f;
    float p0[8], p1[8], corr[8];
#pragma unroll
    for (int r = 0; r < 8; ++r) {
      float a0 = s0[r] * scale, a1 = s1[r] * scale;
      float mx = fmaxf(a0, a1);
#pragma unroll
      for (int msk = 1; msk < 16; msk <<= 1) mx = fmaxf(mx, __shfl_xor(mx, msk, 32));
      float mn = fmaxf(m[r], mx);
      corr[r] = __expf(m[r] - mn);
      m[r] = mn;
      p0[r] = __expf(a0 - mn);
      p1[r] = __expf(a1 - mn);
      float rs = p0[r] + p1[r];
#pragma unroll
      for (int msk = 1; msk < 16; msk <<= 1) rs += __shfl_xor(rs, msk, 32);
      l[r] = l[r] * corr[r] + rs;
    }
#pragma unroll
    for (int n = 0; n < 4; ++n)
#pragma unroll
      for (int r = 0; r < 8; ++r) o[n][r] *= corr[r];

    // P (16x32) -> LDS -> A-fragment
#pragma unroll
    for (int r = 0; r < 8; ++r) {
      Pl[w * 544 + (r + 8 * half) * 34 + l16]      = f2bf(p0[r]);
      Pl[w * 544 + (r + 8 * half) * 34 + 16 + l16] = f2bf(p1[r]);
    }
    __syncthreads();
    FragAB ap;
#pragma unroll
    for (int i = 0; i < 8; ++i) {
      int k0 = (i < 4 ? 2 * i : 16 + 2 * (i - 4)) + 8 * half;
      ap.u[i] = *(const unsigned*)(&Pl[w * 544 + l16 * 34 + k0]);
    }
    // PV: 4 rank tiles of 16
#pragma unroll
    for (int nt = 0; nt < 4; ++nt) {
      FragAB bv;
#pragma unroll
      for (int i = 0; i < 8; ++i) {
        int kp = (i < 4 ? 2 * i : 16 + 2 * (i - 4)) + 8 * half;
        bv.u[i] = *(const unsigned*)(&Vl[(nt * 16 + l16) * 34 + kp]);
      }
      o[nt] = __builtin_amdgcn_wmma_f32_16x16x32_bf16(
          false, ap.v, false, bv.v, (short)0, o[nt], false, false);
    }
    __syncthreads();
  }

#pragma unroll
  for (int nt = 0; nt < 4; ++nt)
#pragma unroll
    for (int r = 0; r < 8; ++r) {
      int row = qw + r + 8 * half;
      int col = nt * 16 + l16;
      float v = o[nt][r] / l[r];
      Oc[((size_t)((b * SEQ + row) * NHEADS + h)) * RANK + col] = f2bf(v);
    }
}

// ---------------------------------------------------------------- gate
__global__ void gate_kernel(const unsigned short* __restrict__ qg,
                            const unsigned short* __restrict__ u,
                            unsigned short* __restrict__ g, int n) {
  int i = blockIdx.x * blockDim.x + threadIdx.x;
  int st = gridDim.x * blockDim.x;
  for (; i < n; i += st) {
    float x = bf2f(qg[i]);
    float s = x / (1.f + __expf(-x));
    g[i] = f2bf(s * bf2f(u[i]));
  }
}

// ---------------------------------------------------------------- host
extern "C" void kernel_launch(void* const* d_in, const int* in_sizes, int n_in,
                              void* d_out, int out_size, void* d_ws, size_t ws_size,
                              hipStream_t stream) {
  (void)in_sizes; (void)n_in; (void)out_size; (void)ws_size;
  const float* hs  = (const float*)d_in[0];
  const int*   pos = (const int*)d_in[1];
  const float* Wq  = (const float*)d_in[2];
  const float* Wk  = (const float*)d_in[3];
  const float* Wv  = (const float*)d_in[4];
  const float* Wkc = (const float*)d_in[5];
  const float* Wvc = (const float*)d_in[6];
  const float* Wqa = (const float*)d_in[7];
  const float* Wqg = (const float*)d_in[8];
  const float* Wov = (const float*)d_in[9];
  const float* Wo  = (const float*)d_in[10];
  float* out = (float*)d_out;

  const int T  = BATCH * SEQ;     // 4096 tokens
  const int TH = T * NHEADS;      // 65536 (token, head) rows

  char* ws = (char*)d_ws;
  size_t cur = 0;
  auto alloc = [&](size_t elems) -> unsigned short* {
    unsigned short* p = (unsigned short*)(ws + cur);
    cur = (cur + elems * sizeof(unsigned short) + 255) & ~(size_t)255;
    return p;
  };
  // transposed bf16 weights (N, K)
  unsigned short* hsb  = alloc((size_t)T * HIDDEN);
  unsigned short* WqT  = alloc((size_t)HIDDEN * HIDDEN);
  unsigned short* WkT  = alloc((size_t)HIDDEN * HIDDEN);
  unsigned short* WvT  = alloc((size_t)HIDDEN * HIDDEN);
  unsigned short* WoT  = alloc((size_t)HIDDEN * HIDDEN);
  unsigned short* WkcT = alloc(HEADD * RANK);
  unsigned short* WvcT = alloc(HEADD * RANK);
  unsigned short* WqaT = alloc(HEADD * RANK);
  unsigned short* WqgT = alloc(HEADD * HEADD);
  unsigned short* WovT = alloc(RANK * HEADD);
  unsigned short* Qf   = alloc((size_t)T * HIDDEN);   // (b,s,h,d) bf16
  unsigned short* Kf   = alloc((size_t)T * HIDDEN);
  unsigned short* Vf   = alloc((size_t)T * HIDDEN);
  unsigned short* Qab  = alloc((size_t)TH * RANK);
  unsigned short* Kcb  = alloc((size_t)TH * RANK);
  unsigned short* Vcb  = alloc((size_t)TH * RANK);
  unsigned short* Qgb  = alloc((size_t)TH * HEADD);
  unsigned short* Ocb  = alloc((size_t)TH * RANK);
  unsigned short* Ub   = alloc((size_t)TH * HEADD);
  unsigned short* Gb   = alloc((size_t)T * HIDDEN);

  auto cvt = [&](const float* s, unsigned short* d, int n) {
    int g = (n + 1023) / 1024; if (g < 1) g = 1; if (g > 16384) g = 16384;
    cvt_f32_bf16<<<g, 256, 0, stream>>>(s, d, n);
  };
  auto cvtT = [&](const float* s, unsigned short* d, int K, int N) {
    cvt_t_f32_bf16<<<dim3(N / 32, K / 32), 256, 0, stream>>>(s, d, K, N);
  };
  auto gemm = [&](const unsigned short* A, const unsigned short* Bt, void* C,
                  int M, int N, int K, bool obf) {
    dim3 g((N + 127) / 128, (M + 127) / 128);
    if (obf) wmma_gemm<1><<<g, 256, 0, stream>>>(A, Bt, C, M, N, K);
    else     wmma_gemm<0><<<g, 256, 0, stream>>>(A, Bt, C, M, N, K);
  };

  // 1) precision conversion (+ weight transpose for async-friendly B tiles)
  cvt(hs, hsb, T * HIDDEN);
  cvtT(Wq,  WqT,  HIDDEN, HIDDEN);
  cvtT(Wk,  WkT,  HIDDEN, HIDDEN);
  cvtT(Wv,  WvT,  HIDDEN, HIDDEN);
  cvtT(Wo,  WoT,  HIDDEN, HIDDEN);
  cvtT(Wkc, WkcT, HEADD, RANK);
  cvtT(Wvc, WvcT, HEADD, RANK);
  cvtT(Wqa, WqaT, HEADD, RANK);
  cvtT(Wqg, WqgT, HEADD, HEADD);
  cvtT(Wov, WovT, RANK, HEADD);

  // 2) QKV projections (bf16 out)
  gemm(hsb, WqT, Qf, T, HIDDEN, HIDDEN, true);
  gemm(hsb, WkT, Kf, T, HIDDEN, HIDDEN, true);
  gemm(hsb, WvT, Vf, T, HIDDEN, HIDDEN, true);

  // 3) RoPE on Q,K in place
  rope_kernel<<<(TH * 32) / 256, 256, 0, stream>>>(Qf, Kf, pos);

  // 4) low-rank projections + gate (heads share weights -> GEMMs over TH rows)
  gemm(Qf, WqaT, Qab, TH, RANK,  HEADD, true);
  gemm(Kf, WkcT, Kcb, TH, RANK,  HEADD, true);
  gemm(Vf, WvcT, Vcb, TH, RANK,  HEADD, true);
  gemm(Qf, WqgT, Qgb, TH, HEADD, HEADD, true);

  // 5) flash attention over rank-64 space
  flash_attn<<<dim3(SEQ / 128, BATCH * NHEADS), 256, 0, stream>>>(Qab, Kcb, Vcb, Ocb);

  // 6) uplift (rank -> head dim)
  gemm(Ocb, WovT, Ub, TH, HEADD, RANK, true);

  // 7) silu(q_gate) * uplift
  {
    int n = T * HIDDEN;
    int g = (n + 1023) / 1024; if (g > 16384) g = 16384;
    gate_kernel<<<g, 256, 0, stream>>>(Qgb, Ub, Gb, n);
  }

  // 8) output projection (fp32 out -> d_out)
  gemm(Gb, WoT, out, T, HIDDEN, HIDDEN, false);
}
